// LocalAttentionLinear_18193481465992
// MI455X (gfx1250) — matrix operand order repaired and checked by reading, not compiled, not run on hardware
//
#include <hip/hip_runtime.h>
#include <hip/hip_bf16.h>

// Shapes (hardcoded per reference): B=4, H=16, N=512, D=E=64, W=256, fp32 I/O.
// Grid: 512 workgroups = (B*H=64) x (N/64=8 row-blocks); 128 threads = 4 wave32.
// Each wave owns a 16-row n-strip; m is looped in 64-wide chunks over the union
// of both masked windows. Matmuls run on v_wmma_f32_16x16x32_f16 (f32 accum).

typedef _Float16 v16h __attribute__((ext_vector_type(16)));
typedef _Float16 v8h  __attribute__((ext_vector_type(8)));
typedef float    v8f  __attribute__((ext_vector_type(8)));

#define LDS_STRIDE 72   // 64 payload halves + 8 pad (keeps 16B alignment, dodges bank conflicts)

// ---- WMMA fragment loaders (gfx1250 wave32 layouts, ISA 7.12.2) ----

// A-matrix 16x32 f16: lane holds row M=lane&15; halves 0..7 -> K=k0+8*hl+i,
// halves 8..15 -> K=k0+16+8*hl+i.  Matrix stored row-major [rows][stride].
__device__ __forceinline__ v16h load_a(const _Float16* buf, int r0, int k0, int lane) {
  const int r  = lane & 15;
  const int hl = lane >> 4;
  const _Float16* p = buf + (r0 + r) * LDS_STRIDE + k0 + 8 * hl;
  v8h lo = *(const v8h*)p;
  v8h hi = *(const v8h*)(p + 16);
  return __builtin_shufflevector(lo, hi, 0,1,2,3,4,5,6,7,8,9,10,11,12,13,14,15);
}

// B-matrix 32x16 f16: lane holds column N=n0+(lane&15); halves 0..7 -> K=k0+8*hl+i,
// halves 8..15 -> K=k0+16+8*hl+i.
// Storage must be "N-major": row index = N, K contiguous within the row.
__device__ __forceinline__ v16h load_b(const _Float16* buf, int n0, int k0, int lane) {
  const int n  = n0 + (lane & 15);
  const int hl = lane >> 4;
  const _Float16* p = buf + n * LDS_STRIDE + k0 + 8 * hl;
  v8h lo = *(const v8h*)p;
  v8h hi = *(const v8h*)(p + 16);
  return __builtin_shufflevector(lo, hi, 0,1,2,3,4,5,6,7,8,9,10,11,12,13,14,15);
}

__device__ __forceinline__ v8f wmma_f16(v16h a, v16h b, v8f c) {
  return __builtin_amdgcn_wmma_f32_16x16x32_f16(false, a, false, b, (short)0, c, false, false);
}

__global__ __launch_bounds__(128)
void lal_wmma_kernel(const float* __restrict__ q,  const float* __restrict__ k,
                     const float* __restrict__ qr, const float* __restrict__ kr,
                     const float* __restrict__ v,  float* __restrict__ out) {
  __shared__ alignas(16) _Float16 sQ [64 * LDS_STRIDE];  // n-block rows, d contiguous
  __shared__ alignas(16) _Float16 sQr[64 * LDS_STRIDE];
  __shared__ alignas(16) _Float16 sK [64 * LDS_STRIDE];  // m-chunk rows, d contiguous (B for S)
  __shared__ alignas(16) _Float16 sKr[64 * LDS_STRIDE];
  __shared__ alignas(16) _Float16 sVT[64 * LDS_STRIDE];  // e rows, m contiguous (B for P@V)
  __shared__ alignas(16) _Float16 sP [4 * 16 * LDS_STRIDE]; // per-wave masked-score strip (A for P@V)

  const int tid  = threadIdx.x;
  const int lane = tid & 31;
  const int w    = tid >> 5;            // wave 0..3 -> n-strip [16w, 16w+16)
  const int bh   = blockIdx.x >> 3;     // 0..63
  const int nb   = blockIdx.x & 7;      // 0..7
  const int n0   = nb * 64;
  const size_t base = (size_t)bh * (512 * 64);

  const int hl   = lane >> 4;
  const int ncol = lane & 15;

  // ---- load Q / Q_rot block once (f32 -> f16) ----
  {
    const int r  = tid >> 1;
    const int c0 = (tid & 1) * 32;
    const float4* gq  = (const float4*)(q  + base + (size_t)(n0 + r) * 64 + c0);
    const float4* gqr = (const float4*)(qr + base + (size_t)(n0 + r) * 64 + c0);
    _Float16* dq  = &sQ [r * LDS_STRIDE + c0];
    _Float16* dqr = &sQr[r * LDS_STRIDE + c0];
#pragma unroll
    for (int i = 0; i < 8; ++i) {
      float4 a = gq[i];
      float4 b = gqr[i];
      dq [4*i+0] = (_Float16)a.x; dq [4*i+1] = (_Float16)a.y;
      dq [4*i+2] = (_Float16)a.z; dq [4*i+3] = (_Float16)a.w;
      dqr[4*i+0] = (_Float16)b.x; dqr[4*i+1] = (_Float16)b.y;
      dqr[4*i+2] = (_Float16)b.z; dqr[4*i+3] = (_Float16)b.w;
    }
  }
  __syncthreads();

  // Q fragments are m-chunk invariant: hoist out of the loop.
  const v16h aq0 = load_a(sQ,  16 * w, 0,  lane);
  const v16h aq1 = load_a(sQ,  16 * w, 32, lane);
  const v16h ar0 = load_a(sQr, 16 * w, 0,  lane);
  const v16h ar1 = load_a(sQr, 16 * w, 32, lane);

  v8f num[4];
#pragma unroll
  for (int t = 0; t < 4; ++t)
#pragma unroll
    for (int j = 0; j < 8; ++j) num[t][j] = 0.0f;

  float denAcc[8];
#pragma unroll
  for (int j = 0; j < 8; ++j) denAcc[j] = 0.0f;

  // Union of win1 and win2 for this row block (uniform across workgroup).
  const int mhi = (n0 < 256) ? (n0 + 319) : (n0 + 63);

  for (int m0 = 0; m0 <= mhi; m0 += 64) {
    __syncthreads();   // previous chunk's LDS reads done before overwrite

    // ---- cooperative load of K / K_rot (row-major) and V (transposed) ----
    {
      const int r  = tid >> 1;
      const int c0 = (tid & 1) * 32;
      const float4* gk  = (const float4*)(k  + base + (size_t)(m0 + r) * 64 + c0);
      const float4* gkr = (const float4*)(kr + base + (size_t)(m0 + r) * 64 + c0);
      const float4* gv  = (const float4*)(v  + base + (size_t)(m0 + r) * 64 + c0);
      _Float16* dk  = &sK [r * LDS_STRIDE + c0];
      _Float16* dkr = &sKr[r * LDS_STRIDE + c0];
#pragma unroll
      for (int i = 0; i < 8; ++i) {
        float4 a = gk[i];
        float4 b = gkr[i];
        float4 c = gv[i];
        dk [4*i+0] = (_Float16)a.x; dk [4*i+1] = (_Float16)a.y;
        dk [4*i+2] = (_Float16)a.z; dk [4*i+3] = (_Float16)a.w;
        dkr[4*i+0] = (_Float16)b.x; dkr[4*i+1] = (_Float16)b.y;
        dkr[4*i+2] = (_Float16)b.z; dkr[4*i+3] = (_Float16)b.w;
        const int e = c0 + 4 * i;
        sVT[(e+0) * LDS_STRIDE + r] = (_Float16)c.x;
        sVT[(e+1) * LDS_STRIDE + r] = (_Float16)c.y;
        sVT[(e+2) * LDS_STRIDE + r] = (_Float16)c.z;
        sVT[(e+3) * LDS_STRIDE + r] = (_Float16)c.w;
      }
    }
    __syncthreads();

    // ---- scores S1 = Q K^T, S2 = Qr Kr^T over the wave's 16x64 strip ----
#pragma unroll
    for (int t = 0; t < 4; ++t) {           // m-subtile [16t, 16t+16)
      v8f c1, c2;
#pragma unroll
      for (int j = 0; j < 8; ++j) { c1[j] = 0.0f; c2[j] = 0.0f; }

      c1 = wmma_f16(aq0, load_b(sK,  16 * t, 0,  lane), c1);
      c1 = wmma_f16(aq1, load_b(sK,  16 * t, 32, lane), c1);
      c2 = wmma_f16(ar0, load_b(sKr, 16 * t, 0,  lane), c2);
      c2 = wmma_f16(ar1, load_b(sKr, 16 * t, 32, lane), c2);

      // mask, accumulate denominator, repack P to f16 in the wave's LDS strip
      const int mg = m0 + 16 * t + ncol;
#pragma unroll
      for (int j = 0; j < 8; ++j) {
        const int ng = n0 + 16 * w + j + 8 * hl;
        const bool in1 = (mg <= ng) && (mg + 256 > ng);                      // causal 256-window
        const bool in2 = (ng < 256) ? ((mg > ng) && (mg <= ng + 256))        // quirky rot window
                                    : (mg <= ng);                            // raw cumsum tail
        const float s1 = c1[j];
        const float s2 = c2[j];
        const float pv = (in1 ? s1 : 0.0f) + (in2 ? s2 : 0.0f);
        denAcc[j] += in1 ? (s1 + s2) : 0.0f;
        sP[(w * 16 + j + 8 * hl) * LDS_STRIDE + 16 * t + ncol] = (_Float16)pv;
      }
    }

    // wave-private LDS strip: DS ops are in-order per wave; block compiler reordering
    asm volatile("s_wait_dscnt 0" ::: "memory");

    // ---- num += P @ V  (16x64 strip @ 64x64) ----
    const v16h ap0 = load_a(sP, 16 * w, 0,  lane);
    const v16h ap1 = load_a(sP, 16 * w, 32, lane);
#pragma unroll
    for (int te = 0; te < 4; ++te) {        // e-subtile [16te, 16te+16)
      num[te] = wmma_f16(ap0, load_b(sVT, 16 * te, 0,  lane), num[te]);
      num[te] = wmma_f16(ap1, load_b(sVT, 16 * te, 32, lane), num[te]);
    }
  }

  // ---- denominator: reduce over the 16 lanes holding row M=j+8*hl ----
#pragma unroll
  for (int j = 0; j < 8; ++j) {
    float d = denAcc[j];
    d += __shfl_xor(d, 1, 32);
    d += __shfl_xor(d, 2, 32);
    d += __shfl_xor(d, 4, 32);
    d += __shfl_xor(d, 8, 32);
    denAcc[j] = 1.0f / (d + 1e-12f);
  }

  // ---- scale + store (C layout: M=j+8*hl, N=lane&15) ----
#pragma unroll
  for (int te = 0; te < 4; ++te) {
#pragma unroll
    for (int j = 0; j < 8; ++j) {
      const int ng = n0 + 16 * w + j + 8 * hl;
      out[base + (size_t)ng * 64 + 16 * te + ncol] = num[te][j] * denAcc[j];
    }
  }
}

extern "C" void kernel_launch(void* const* d_in, const int* in_sizes, int n_in,
                              void* d_out, int out_size, void* d_ws, size_t ws_size,
                              hipStream_t stream) {
  (void)in_sizes; (void)n_in; (void)out_size; (void)d_ws; (void)ws_size;
  const float* q  = (const float*)d_in[0];
  const float* k  = (const float*)d_in[1];
  const float* qr = (const float*)d_in[2];
  const float* kr = (const float*)d_in[3];
  const float* v  = (const float*)d_in[4];
  float* out = (float*)d_out;
  // 64 (b,h) pairs x 8 row-blocks of 64 = 512 workgroups, 4 wave32 each.
  lal_wmma_kernel<<<dim3(512), dim3(128), 0, stream>>>(q, k, qr, kr, v, out);
}
